// GravitonFeedForward_15719580303660
// MI455X (gfx1250) — matrix operand order repaired and checked by reading, not compiled, not run on hardware
//
#include <hip/hip_runtime.h>

// ---------------------------------------------------------------------------
// GravitonFeedForward for MI455X (gfx1250, wave32, WMMA bf16)
//   B=2, S=4096 -> BS=8192 tokens; H=2048; I=8192; K=4096 (= I/2 kept)
//   gate = x @ Wg^T ; thr = kth largest |gate| per token ; up = x @ Wu^T
//   out  = ((|gate|>=thr ? gate : 0) * up) @ Wd^T
// ~8.25e11 FLOP vs ~320MB compulsory traffic (AI ~2400 FLOP/B) => WMMA bf16.
// All bf16 operands (x + 3 weights = 128MB) fit in the 192MB L2 together, so
// B-operands stream from L2; block-shared A panels are staged into LDS via
// CDNA5 async global->LDS copies (ASYNCcnt) with double buffering.
// ---------------------------------------------------------------------------

#define BS_TOK  8192
#define H_DIM   2048
#define I_DIM   8192
#define K_KEEP  4096

typedef __attribute__((ext_vector_type(16))) __bf16 v16bf;
typedef __attribute__((ext_vector_type(8)))  float  v8f;
typedef __attribute__((ext_vector_type(4)))  unsigned int u32x4;
typedef __attribute__((ext_vector_type(4)))  int i32x4;

// async-copy operand types: v4i in global (AS1) / LDS (AS3) address spaces
typedef __attribute__((address_space(1))) i32x4 g_i32x4;
typedef __attribute__((address_space(3))) i32x4 l_i32x4;

struct Frag32 { u32x4 lo, hi; };   // 32 bytes == one v16bf operand

// ---- CDNA5 async global->LDS staging (guarded; sync-copy fallback) --------
#if __has_builtin(__builtin_amdgcn_global_load_async_to_lds_b128) && \
    __has_builtin(__builtin_amdgcn_s_wait_asynccnt)
#define ASYNC_STAGE16(gp, lp)                                                  \
    __builtin_amdgcn_global_load_async_to_lds_b128(                            \
        reinterpret_cast<g_i32x4*>((unsigned long long)(gp)),                  \
        reinterpret_cast<l_i32x4*>((unsigned)(unsigned long long)(lp)), 0, 0)
#define ASYNC_WAIT1() __builtin_amdgcn_s_wait_asynccnt(1)
#define ASYNC_WAIT0() __builtin_amdgcn_s_wait_asynccnt(0)
#else
#define ASYNC_STAGE16(gp, lp) (*(u32x4*)(lp) = *(const u32x4*)(gp))
#define ASYNC_WAIT1()
#define ASYNC_WAIT0()
#endif

static __device__ __forceinline__ unsigned short f2bf(float f) {
    unsigned u = __builtin_bit_cast(unsigned, f);
    unsigned r = u + 0x7FFFu + ((u >> 16) & 1u);   // round-to-nearest-even
    return (unsigned short)(r >> 16);
}

// ---------------------------------------------------------------------------
// f32 -> bf16 conversion
// ---------------------------------------------------------------------------
__global__ __launch_bounds__(256)
void cvt_f32_bf16_kernel(const float* __restrict__ in,
                         unsigned short* __restrict__ out, int n) {
    int i = blockIdx.x * 256 + threadIdx.x;
    if (i < n) out[i] = f2bf(in[i]);
}

// ---------------------------------------------------------------------------
// Register-blocked WMMA GEMM:  C[m,n] = sum_k A[m,k] * B[n,k]
//   (A,B row-major bf16; C f32). NB=2 computes two outputs (g,u) sharing A.
//
// Block = 256 thr = 8 waves. Block tile: M=32 (2 wave m-tiles shared by all
// waves), N = 8*NT*16 per block (each wave owns NT n-tiles). K-chunks of 64:
// the A panel (32x64 bf16 = 4KB) is staged into LDS by the whole block, one
// async b128 per thread, double-buffered (ASYNCcnt + barrier), then read back
// as WMMA fragments via ds_load_b128. B fragments stream from L2 directly.
// Per wave, per 64-K chunk: 2 k-steps x (2 MT x NT x NB) WMMAs from
// 4*NT*NB global b128 + 8 LDS b128 -> ~2x the FLOP/B of a 1-tile wave.
//
// Fragment layouts per CDNA5 ISA 7.12.2 (wave32):
//  A 16x32 bf16 : lane L -> row M=L&15 ; elems 0..7 = K [g..g+7],
//                 elems 8..15 = K [16+g..23+g], g = (L>>4)*8
//  B 32x16 bf16 : lane L -> col N=L&15 ; elems 0..7 = K [g..g+7],
//                 elems 8..15 = K [16+g..23+g], g = (L>>4)*8
//  C 16x16 f32  : VGPR r, lane L -> M = r + 8*(L>>4), N = L&15
// ---------------------------------------------------------------------------
template <int NT, int NB>
__global__ __launch_bounds__(256)
void wmma_gemm_nt_kernel(const unsigned short* __restrict__ A,
                         const unsigned short* __restrict__ B0,
                         const unsigned short* __restrict__ B1,
                         float* __restrict__ C0,
                         float* __restrict__ C1,
                         int Kdim, int ldc) {
    __shared__ unsigned short panel[2][32 * 64];   // 8KB double-buffered A

    const int tid  = threadIdx.x;
    const int lane = tid & 31;
    const int wave = tid >> 5;
    const int m0   = blockIdx.x * 32;
    const int n0   = blockIdx.y * (8 * NT * 16) + wave * (NT * 16);

    // A-panel staging: thread t moves bytes [t*16, t*16+16) of each chunk.
    const int sRow = tid >> 3;                      // 0..31
    const int sCol = (tid & 7) << 3;                // element col, step 8
    const unsigned short* aSrc = A + (size_t)(m0 + sRow) * Kdim + sCol;

    // B fragment base pointers (k=0)
    const unsigned short* bPtr[NT][NB];
#pragma unroll
    for (int nt = 0; nt < NT; ++nt)
#pragma unroll
        for (int nb = 0; nb < NB; ++nb)
            bPtr[nt][nb] = (nb ? B1 : B0)
                + (size_t)(n0 + nt * 16 + (lane & 15)) * Kdim
                + ((lane >> 4) << 3);

    // A fragment LDS element offsets (panel row-major [32][64])
    int aOff[2];
#pragma unroll
    for (int mt = 0; mt < 2; ++mt)
        aOff[mt] = (mt * 16 + (lane & 15)) * 64 + ((lane >> 4) << 3);

    v8f acc[2][NT][NB] = {};

    const int nChunks = Kdim >> 6;                  // K-chunks of 64
    ASYNC_STAGE16(aSrc, &panel[0][tid << 3]);       // prologue: chunk 0

    for (int c = 0; c < nChunks; ++c) {
        if (c + 1 < nChunks) {
            ASYNC_STAGE16(aSrc + (size_t)(c + 1) * 64,
                          &panel[(c + 1) & 1][tid << 3]);
            ASYNC_WAIT1();                          // chunk c landed (this wave)
        } else {
            ASYNC_WAIT0();
        }
        __syncthreads();                            // panel visible block-wide

        const unsigned short* p = panel[c & 1];
        const int koff = c << 6;
#pragma unroll
        for (int kk = 0; kk < 64; kk += 32) {
            // A fragments from LDS (ds_load_b128 x2 per m-tile)
            v16bf afrag[2];
#pragma unroll
            for (int mt = 0; mt < 2; ++mt) {
                Frag32 fa;
                fa.lo = *(const u32x4*)(p + aOff[mt] + kk);
                fa.hi = *(const u32x4*)(p + aOff[mt] + kk + 16);
                afrag[mt] = __builtin_bit_cast(v16bf, fa);
            }
            // B fragments from global (L2-resident weights)
            v16bf bfrag[NT][NB];
#pragma unroll
            for (int nt = 0; nt < NT; ++nt)
#pragma unroll
                for (int nb = 0; nb < NB; ++nb) {
                    const unsigned short* bp = bPtr[nt][nb] + koff + kk;
                    __builtin_prefetch(bp + 256, 0, 1);
                    Frag32 fb;
                    fb.lo = *(const u32x4*)(bp);
                    fb.hi = *(const u32x4*)(bp + 16);
                    bfrag[nt][nb] = __builtin_bit_cast(v16bf, fb);
                }
#pragma unroll
            for (int mt = 0; mt < 2; ++mt)
#pragma unroll
                for (int nt = 0; nt < NT; ++nt)
#pragma unroll
                    for (int nb = 0; nb < NB; ++nb)
                        acc[mt][nt][nb] = __builtin_amdgcn_wmma_f32_16x16x32_bf16(
                            false, afrag[mt], false, bfrag[nt][nb],
                            (short)0, acc[mt][nt][nb], false, false);
        }
        __syncthreads();          // all reads done before next buffer overwrite
    }

    const int cRow = m0 + ((lane >> 4) << 3);
    const int cCol = n0 + (lane & 15);
#pragma unroll
    for (int mt = 0; mt < 2; ++mt)
#pragma unroll
        for (int nt = 0; nt < NT; ++nt)
#pragma unroll
            for (int nb = 0; nb < NB; ++nb) {
                float* C = nb ? C1 : C0;
#pragma unroll
                for (int r = 0; r < 8; ++r)
                    C[(size_t)(cRow + mt * 16 + r) * ldc + cCol + nt * 16] =
                        acc[mt][nt][nb][r];
            }
}

// ---------------------------------------------------------------------------
// Per-token k-th-largest-|g| threshold: one block per token row.
// Row's |g| bit-keys cached in LDS (32KB of the 320KB WGP LDS), then a
// 4-pass MSD radix select (256-bin LDS histogram per pass).
// Matches reference tie semantics: mask keeps all values with |g| >= thr.
// ---------------------------------------------------------------------------
__global__ __launch_bounds__(256)
void topk_thresh_kernel(const float* __restrict__ g, float* __restrict__ thr) {
    __shared__ unsigned keys[I_DIM];
    __shared__ unsigned hist[256];
    __shared__ unsigned sPrefix, sWant;

    const int row = blockIdx.x;
    const int tid = threadIdx.x;
    const float* grow = g + (size_t)row * I_DIM;

    for (int i = tid; i < I_DIM; i += 256)
        keys[i] = __builtin_bit_cast(unsigned, grow[i]) & 0x7FFFFFFFu;
    if (tid == 0) { sPrefix = 0u; sWant = K_KEEP; }
    __syncthreads();

    for (int shift = 24; shift >= 0; shift -= 8) {
        hist[tid] = 0u;
        __syncthreads();
        const unsigned pfx  = sPrefix;
        const unsigned mask = (shift == 24) ? 0u : (0xFFFFFFFFu << (shift + 8));
        for (int i = tid; i < I_DIM; i += 256) {
            unsigned k = keys[i];
            if ((k & mask) == pfx)
                atomicAdd(&hist[(k >> shift) & 0xFFu], 1u);
        }
        __syncthreads();
        if (tid == 0) {
            unsigned want = sWant, acc = 0u;
            for (int b = 255; b >= 0; --b) {
                unsigned h = hist[b];
                if (acc + h >= want) {
                    sWant   = want - acc;
                    sPrefix = pfx | ((unsigned)b << shift);
                    break;
                }
                acc += h;
            }
        }
        __syncthreads();
    }
    if (tid == 0) thr[row] = __builtin_bit_cast(float, sPrefix);
}

// ---------------------------------------------------------------------------
// a = mask(g) * u, converted to bf16 for the final WMMA GEMM
// ---------------------------------------------------------------------------
__global__ __launch_bounds__(256)
void mask_mul_kernel(const float* __restrict__ g, const float* __restrict__ u,
                     const float* __restrict__ thr,
                     unsigned short* __restrict__ a) {
    long long idx = (long long)blockIdx.x * 256 + threadIdx.x;
    if (idx >= (long long)BS_TOK * I_DIM) return;
    int row  = (int)(idx >> 13);           // / I_DIM (8192)
    float gv = g[idx];
    float r  = (fabsf(gv) >= thr[row]) ? gv * u[idx] : 0.0f;
    a[idx] = f2bf(r);
}

// ---------------------------------------------------------------------------
extern "C" void kernel_launch(void* const* d_in, const int* in_sizes, int n_in,
                              void* d_out, int out_size, void* d_ws, size_t ws_size,
                              hipStream_t stream) {
    const float* x  = (const float*)d_in[0];   // [BS, H]
    const float* Wg = (const float*)d_in[1];   // [I, H]
    const float* Wu = (const float*)d_in[2];   // [I, H]
    const float* Wd = (const float*)d_in[3];   // [H, I]
    float* out = (float*)d_out;                // [BS, H]

    // workspace layout (all sizes 256B-aligned by construction)
    char* ws = (char*)d_ws;
    const size_t szXH = (size_t)BS_TOK * H_DIM * 2;   // 32MB bf16 x
    const size_t szW  = (size_t)I_DIM  * H_DIM * 2;   // 32MB per bf16 weight
    const size_t szGU = (size_t)BS_TOK * I_DIM * 4;   // 256MB f32 g / u
    const size_t szA  = (size_t)BS_TOK * I_DIM * 2;   // 128MB bf16 activation

    unsigned short* xh  = (unsigned short*)(ws);
    unsigned short* wgh = (unsigned short*)(ws + szXH);
    unsigned short* wuh = (unsigned short*)(ws + szXH + szW);
    unsigned short* wdh = (unsigned short*)(ws + szXH + 2 * szW);
    float*          g   = (float*)         (ws + szXH + 3 * szW);
    float*          u   = (float*)         (ws + szXH + 3 * szW + szGU);
    unsigned short* ah  = (unsigned short*)(ws + szXH + 3 * szW + 2 * szGU);
    float*          thr = (float*)         (ws + szXH + 3 * szW + 2 * szGU + szA);

    // 1) f32 -> bf16 conversions
    {
        int nX = BS_TOK * H_DIM, nW = I_DIM * H_DIM;
        cvt_f32_bf16_kernel<<<(nX + 255) / 256, 256, 0, stream>>>(x,  xh,  nX);
        cvt_f32_bf16_kernel<<<(nW + 255) / 256, 256, 0, stream>>>(Wg, wgh, nW);
        cvt_f32_bf16_kernel<<<(nW + 255) / 256, 256, 0, stream>>>(Wu, wuh, nW);
        cvt_f32_bf16_kernel<<<(nW + 255) / 256, 256, 0, stream>>>(Wd, wdh, nW);
    }

    // 2) fused gate+up GEMM: g/u = x @ {Wg,Wu}^T   (K over H; NT=2, NB=2)
    //    block tile 32 x 256 (of I), grid (8192/32, 8192/256)
    wmma_gemm_nt_kernel<2, 2>
        <<<dim3(BS_TOK / 32, I_DIM / 256), 256, 0, stream>>>(
            xh, wgh, wuh, g, u, H_DIM, I_DIM);

    // 3) per-token k-th largest |gate| threshold
    topk_thresh_kernel<<<BS_TOK, 256, 0, stream>>>(g, thr);

    // 4) a = mask(g) * u  -> bf16
    {
        long long total = (long long)BS_TOK * I_DIM;
        mask_mul_kernel<<<(unsigned)((total + 255) / 256), 256, 0, stream>>>(
            g, u, thr, ah);
    }

    // 5) out = a @ Wd^T   (K over I; NT=4, NB=1)
    //    block tile 32 x 512 (of H), grid (8192/32, 2048/512)
    wmma_gemm_nt_kernel<4, 1>
        <<<dim3(BS_TOK / 32, H_DIM / 512), 256, 0, stream>>>(
            ah, wdh, nullptr, out, nullptr, I_DIM, H_DIM);
}
